// LessSimpleMoE_55679956025636
// MI455X (gfx1250) — compile-verified, matrix operand order checked by reading, not run on hardware
//
#include <hip/hip_runtime.h>
#include <hip/hip_bf16.h>

// Problem constants (match reference).
#define Nn   16384
#define Dd   1024
#define Ee   5
#define Mm   64
#define MODS 5

#define BM 64
#define BN 128
#define BK 32
#define KSTEPS (Dd / BK)
#define LDA 18   // dword stride: 16 data dwords + 2 pad (18m mod 64 distinct for m<16, rows 8B-aligned)

typedef __attribute__((ext_vector_type(16))) __bf16 v16bf;
typedef __attribute__((ext_vector_type(2)))  __bf16 v2bf;
typedef __attribute__((ext_vector_type(2)))  float  v2f;
typedef __attribute__((ext_vector_type(8)))  float  v8f;

union FragAB { unsigned u[8]; v16bf v; };

// f32 pair -> packed bf16 dword: single v_cvt_pk_bf16_f32.
__device__ __forceinline__ unsigned pack_bf16(float a, float b) {
  v2f f = {a, b};
  v2bf h = __builtin_convertvector(f, v2bf);
  return __builtin_bit_cast(unsigned, h);
}

// ---------------------------------------------------------------------------
// Router: one wave32 per row. Computes 0.5*final_w[n,e] and 0.5*gates[n,q].
// ---------------------------------------------------------------------------
__global__ __launch_bounds__(256)
void router_kernel(const float* __restrict__ x,
                   const float* __restrict__ Wsoft,  const float* __restrict__ bsoft,
                   const float* __restrict__ Wsparse,const float* __restrict__ bsparse,
                   const float* __restrict__ m_key,  const float* __restrict__ m_val,
                   const float* __restrict__ Wg,     const float* __restrict__ bg,
                   float* __restrict__ fw_s, float* __restrict__ g_s) {
  const int lane = threadIdx.x & 31;
  const int wid  = threadIdx.x >> 5;
  const int n    = blockIdx.x * 8 + wid;

  const float* xr = x + (long)n * Dd;
  float xv[32];
#pragma unroll
  for (int i = 0; i < 32; ++i) xv[i] = xr[lane + 32 * i];

  auto wdot = [&](const float* __restrict__ w) -> float {
    float s = 0.f;
#pragma unroll
    for (int i = 0; i < 32; ++i) s += xv[i] * w[lane + 32 * i];
#pragma unroll
    for (int off = 16; off >= 1; off >>= 1) s += __shfl_xor(s, off, 32);
    return s;  // full dot, same value in all lanes
  };

  float softl[Ee], sparsel[Ee], gatev[MODS];
#pragma unroll
  for (int e = 0; e < Ee; ++e) softl[e]   = wdot(Wsoft   + e * Dd) + bsoft[e];
#pragma unroll
  for (int e = 0; e < Ee; ++e) sparsel[e] = wdot(Wsparse + e * Dd) + bsparse[e];
#pragma unroll
  for (int q = 0; q < MODS; ++q) gatev[q] = wdot(Wg + q * Dd) + bg[q];

  // memory attention: attn = softmax(x @ m_key.T / 32)
  float attnv[2] = {0.f, 0.f};
  for (int m = 0; m < Mm; ++m) {
    float s = wdot(m_key + m * Dd) * 0.03125f;  // 1/sqrt(1024)
    if (lane == (m & 31)) attnv[m >> 5] = s;
  }
  float mx = fmaxf(attnv[0], attnv[1]);
#pragma unroll
  for (int off = 16; off >= 1; off >>= 1) mx = fmaxf(mx, __shfl_xor(mx, off, 32));
  attnv[0] = __expf(attnv[0] - mx);
  attnv[1] = __expf(attnv[1] - mx);
  float ssum = attnv[0] + attnv[1];
#pragma unroll
  for (int off = 16; off >= 1; off >>= 1) ssum += __shfl_xor(ssum, off, 32);
  float inv = 1.f / ssum;
  attnv[0] *= inv; attnv[1] *= inv;

  float meml[Ee];
#pragma unroll
  for (int e = 0; e < Ee; ++e) {
    float p = attnv[0] * m_val[lane * Ee + e] + attnv[1] * m_val[(lane + 32) * Ee + e];
#pragma unroll
    for (int off = 16; off >= 1; off >>= 1) p += __shfl_xor(p, off, 32);
    meml[e] = p;
  }

  float mxa = softl[0], mxb = meml[0];
#pragma unroll
  for (int e = 1; e < Ee; ++e) { mxa = fmaxf(mxa, softl[e]); mxb = fmaxf(mxb, meml[e]); }
  float sa = 0.f, sb = 0.f;
#pragma unroll
  for (int e = 0; e < Ee; ++e) {
    softl[e] = __expf(softl[e] - mxa); sa += softl[e];
    meml[e]  = __expf(meml[e]  - mxb); sb += meml[e];
  }
  sa = 1.f / sa; sb = 1.f / sb;

  int amax = 0; float mv = sparsel[0];
#pragma unroll
  for (int e = 1; e < Ee; ++e) if (sparsel[e] > mv) { mv = sparsel[e]; amax = e; }

  if (lane == 0) {
#pragma unroll
    for (int e = 0; e < Ee; ++e) {
      float sp = (e == amax) ? 1.f : 0.f;
      float fin = 0.5f * (0.5f * sp + 0.5f * softl[e] * sa) + 0.5f * (meml[e] * sb);
      fw_s[n * Ee + e] = 0.5f * fin;          // pre-scale by (1-alpha_fusion)
    }
#pragma unroll
    for (int q = 0; q < MODS; ++q)
      g_s[n * MODS + q] = 0.5f / (1.f + __expf(-gatev[q]));  // pre-scale by alpha_fusion
  }
}

// ---------------------------------------------------------------------------
// out[n, o] = 0.5 * bf[o]   (fusion bias applied once)
// ---------------------------------------------------------------------------
__global__ void init_out(float* __restrict__ out, const float* __restrict__ bf) {
  long i = (long)blockIdx.x * 256 + threadIdx.x;
  out[i] = 0.5f * bf[i & (Dd - 1)];
}

// ---------------------------------------------------------------------------
// Tiled bf16 WMMA GEMM: C[m,o] = sum_d A[m,d] * B[o,d]  (B row-major, stride ldb)
// MODE 0: H[m,o] = bf16(silu(C + bias[o]))            (expert hidden)
// MODE 1: out[m,o] += scale[m*5+sidx] * (C + bias?)   (weighted accumulate)
// ABF16: A is bf16 (H) -> staged via global_load_async_to_lds (ASYNCcnt path);
// otherwise A is f32, staged through registers with v_cvt_pk_bf16_f32.
// 8 waves, block tile 64x128, wave tile 32x32, double-buffered LDS.
// ---------------------------------------------------------------------------
template <int MODE, bool ABF16>
__global__ __launch_bounds__(256)
void gemm_bf16_kernel(const float* __restrict__ Af, const __hip_bfloat16* __restrict__ Ah,
                      const float* __restrict__ B, long ldb,
                      const float* __restrict__ bias,
                      const float* __restrict__ scale, int sidx,
                      float* __restrict__ outF, __hip_bfloat16* __restrict__ outH) {
  __shared__ unsigned sA[2][BM * LDA];
  __shared__ unsigned sB[2][BN * LDA];

  const int tid  = threadIdx.x;
  const int lane = tid & 31;
  const int wid  = tid >> 5;
  const int wm   = wid & 1;          // 2 wave rows
  const int wn   = wid >> 1;         // 4 wave cols
  const long bm  = (long)blockIdx.x * BM;
  const long bn  = (long)blockIdx.y * BN;

  // staging coordinates
  const int ar = tid >> 2, ac = (tid & 3) * 8;   // A: 8 elems/thread
  const int br = tid >> 1, bc = (tid & 1) * 16;  // B: 16 f32/thread

  float4 ra[2];
  float4 rb[4];

  auto issueA = [&](int kb, int buf) {
    if constexpr (ABF16) {
      unsigned l0 = (unsigned)(size_t)&sA[buf][ar * LDA + (ac >> 1)];
      unsigned long g = (unsigned long)(size_t)(Ah + (bm + ar) * (long)Dd + kb + ac);
      asm volatile("global_load_async_to_lds_b64 %0, %1, off" :: "v"(l0), "v"(g) : "memory");
      asm volatile("global_load_async_to_lds_b64 %0, %1, off" :: "v"(l0 + 8u), "v"(g + 8u) : "memory");
    } else {
      const float4* ap = (const float4*)(Af + (bm + ar) * (long)Dd + kb + ac);
      ra[0] = ap[0];
      ra[1] = ap[1];
    }
  };
  auto commitA = [&](int buf) {
    if constexpr (!ABF16) {
      unsigned* d = &sA[buf][ar * LDA + (ac >> 1)];
      *(uint2*)(d + 0) = make_uint2(pack_bf16(ra[0].x, ra[0].y), pack_bf16(ra[0].z, ra[0].w));
      *(uint2*)(d + 2) = make_uint2(pack_bf16(ra[1].x, ra[1].y), pack_bf16(ra[1].z, ra[1].w));
    }
  };
  auto issueB = [&](int kb) {
    const float4* bp = (const float4*)(B + (bn + br) * ldb + kb + bc);
#pragma unroll
    for (int i = 0; i < 4; ++i) rb[i] = bp[i];
  };
  auto commitB = [&](int buf) {
    unsigned* d = &sB[buf][br * LDA + (bc >> 1)];
#pragma unroll
    for (int i = 0; i < 4; ++i)
      *(uint2*)(d + 2 * i) = make_uint2(pack_bf16(rb[i].x, rb[i].y), pack_bf16(rb[i].z, rb[i].w));
  };

  // ---- prologue: stage tile 0 ----
  issueA(0, 0);
  issueB(0);
  commitA(0);
  commitB(0);
  if constexpr (ABF16) asm volatile("s_wait_asynccnt 0x0" ::: "memory");
  __syncthreads();

  v8f acc[2][2] = {};
  const int lr   = lane & 15;
  const int koff = (lane >> 4) << 2;  // lanes 0-15: K 0..7 / 16..23, lanes 16-31: K 8..15 / 24..31

  for (int kt = 0; kt < KSTEPS; ++kt) {
    const int cur = kt & 1, nxt = cur ^ 1;
    const bool more = (kt + 1) < KSTEPS;
    if (more) { issueA((kt + 1) * BK, nxt); issueB((kt + 1) * BK); }
    if (kt + 2 < KSTEPS) {  // distance-2 L2 prefetch (global_prefetch_b8)
      long ga = (bm + (tid & 63)) * (long)Dd + (kt + 2) * BK;
      if constexpr (ABF16) __builtin_prefetch((const void*)(Ah + ga), 0, 1);
      else                 __builtin_prefetch((const void*)(Af + ga), 0, 1);
      if (tid < BN) __builtin_prefetch((const void*)(B + (bn + tid) * ldb + (kt + 2) * BK), 0, 1);
    }

    // ---- fragment loads (ISA 16-bit 16x32 layout) + WMMA ----
    FragAB a[2], b[2];
#pragma unroll
    for (int i = 0; i < 2; ++i) {
      const unsigned* s = &sA[cur][(wm * 32 + i * 16 + lr) * LDA + koff];
      uint2 q0 = *(const uint2*)(s + 0), q1 = *(const uint2*)(s + 2);
      uint2 q2 = *(const uint2*)(s + 8), q3 = *(const uint2*)(s + 10);
      a[i].u[0] = q0.x; a[i].u[1] = q0.y; a[i].u[2] = q1.x; a[i].u[3] = q1.y;
      a[i].u[4] = q2.x; a[i].u[5] = q2.y; a[i].u[6] = q3.x; a[i].u[7] = q3.y;
    }
#pragma unroll
    for (int i = 0; i < 2; ++i) {
      const unsigned* s = &sB[cur][(wn * 32 + i * 16 + lr) * LDA + koff];
      uint2 q0 = *(const uint2*)(s + 0), q1 = *(const uint2*)(s + 2);
      uint2 q2 = *(const uint2*)(s + 8), q3 = *(const uint2*)(s + 10);
      b[i].u[0] = q0.x; b[i].u[1] = q0.y; b[i].u[2] = q1.x; b[i].u[3] = q1.y;
      b[i].u[4] = q2.x; b[i].u[5] = q2.y; b[i].u[6] = q3.x; b[i].u[7] = q3.y;
    }
#pragma unroll
    for (int i = 0; i < 2; ++i)
#pragma unroll
      for (int j = 0; j < 2; ++j)
        acc[i][j] = __builtin_amdgcn_wmma_f32_16x16x32_bf16(
            false, a[i].v, false, b[j].v, (short)0, acc[i][j], false, false);

    if (more) { commitA(nxt); commitB(nxt); }
    if constexpr (ABF16) asm volatile("s_wait_asynccnt 0x0" ::: "memory");
    __syncthreads();
  }

  // ---- epilogue: C layout = VGPR v, lanes 0-15 -> M=v, lanes 16-31 -> M=v+8 ----
#pragma unroll
  for (int i = 0; i < 2; ++i) {
#pragma unroll
    for (int j = 0; j < 2; ++j) {
      long row0 = bm + wm * 32 + i * 16 + ((lane >> 4) << 3);
      long col  = bn + wn * 32 + j * 16 + (lane & 15);
      float bv = bias ? bias[col] : 0.f;
#pragma unroll
      for (int v = 0; v < 8; ++v) {
        long row = row0 + v;
        float val = acc[i][j][v] + bv;
        long idx = row * Dd + col;
        if constexpr (MODE == 0) {
          float s = val / (1.f + __expf(-val));          // SiLU
          outH[idx] = __float2bfloat16(s);
        } else {
          float sc = scale[row * 5 + sidx];
          outF[idx] = outF[idx] + sc * val;
        }
      }
    }
  }
}

// ---------------------------------------------------------------------------
extern "C" void kernel_launch(void* const* d_in, const int* in_sizes, int n_in,
                              void* d_out, int out_size, void* d_ws, size_t ws_size,
                              hipStream_t stream) {
  const float* x       = (const float*)d_in[0];
  const float* feats[5] = {(const float*)d_in[1], (const float*)d_in[2],
                           (const float*)d_in[3], (const float*)d_in[4],
                           (const float*)d_in[5]};
  const float* Wsoft   = (const float*)d_in[6];
  const float* bsoft   = (const float*)d_in[7];
  const float* Wsparse = (const float*)d_in[8];
  const float* bsparse = (const float*)d_in[9];
  const float* m_key   = (const float*)d_in[10];
  const float* m_val   = (const float*)d_in[11];
  const float* W1      = (const float*)d_in[12];
  const float* b1      = (const float*)d_in[13];
  const float* W2      = (const float*)d_in[14];
  const float* b2      = (const float*)d_in[15];
  const float* Wg      = (const float*)d_in[16];
  const float* bg      = (const float*)d_in[17];
  const float* Wf      = (const float*)d_in[18];
  const float* bf      = (const float*)d_in[19];
  float* out = (float*)d_out;

  // Workspace: 0.5*final_w [N,5] f32 | 0.5*gates [N,5] f32 | H [N,D] bf16  (~34.2 MB)
  float* fw_s = (float*)d_ws;
  float* g_s  = fw_s + (size_t)Nn * Ee;
  __hip_bfloat16* H = (__hip_bfloat16*)(g_s + (size_t)Nn * MODS);

  router_kernel<<<Nn / 8, 256, 0, stream>>>(x, Wsoft, bsoft, Wsparse, bsparse,
                                            m_key, m_val, Wg, bg, fw_s, g_s);
  init_out<<<(int)(((long)Nn * Dd) / 256), 256, 0, stream>>>(out, bf);

  dim3 grid(Nn / BM, Dd / BN);
  for (int e = 0; e < Ee; ++e) {
    // H = silu(x @ W1[e].T + b1[e])   (bf16)
    gemm_bf16_kernel<0, false><<<grid, 256, 0, stream>>>(
        x, nullptr, W1 + (size_t)e * Dd * Dd, (long)Dd, b1 + (size_t)e * Dd,
        nullptr, 0, nullptr, H);
    // out += 0.5*final_w[:,e] * (H @ W2[e].T + b2[e])   (async A staging)
    gemm_bf16_kernel<1, true><<<grid, 256, 0, stream>>>(
        nullptr, H, W2 + (size_t)e * Dd * Dd, (long)Dd, b2 + (size_t)e * Dd,
        fw_s, e, out, nullptr);
  }
  for (int q = 0; q < MODS; ++q) {
    // out += 0.5*gates[:,q] * (feat_q @ Wf[:, qD:(q+1)D].T)
    gemm_bf16_kernel<1, false><<<grid, 256, 0, stream>>>(
        feats[q], nullptr, Wf + (size_t)q * Dd, (long)(MODS * Dd), nullptr,
        g_s, q, out, nullptr);
  }
}
